// CausalAnalysisHierPredictor_88871463289477
// MI455X (gfx1250) — matrix operand order repaired and checked by reading, not compiled, not run on hardware
//
#include <hip/hip_runtime.h>
#include <hip/hip_bf16.h>
#include <math.h>

#define HID   512
#define POOL  4096
#define NOBJ  151
#define NREL  51
#define N_OBJS  8192
#define N_PAIRS 32768

typedef __attribute__((ext_vector_type(16))) __bf16 v16bf;
typedef __attribute__((ext_vector_type(8)))  __bf16 v8bf;
typedef __attribute__((ext_vector_type(8)))  float  v8f;
typedef __attribute__((ext_vector_type(4)))  int    i4v;

// Async global->LDS DMA path (ASYNCcnt), if this toolchain declares the builtin.
#if defined(__HIP_DEVICE_COMPILE__)
#if __has_builtin(__builtin_amdgcn_global_load_async_to_lds_b128)
#define HAS_ASYNC_LDS 1
#endif
#endif
#ifndef HAS_ASYNC_LDS
#define HAS_ASYNC_LDS 0
#endif

// ---------------- workspace layout (byte offsets into d_ws) ----------------
#define OFF_WEMBT  ((size_t)0x00000000)  // bf16 [1024][512]   (N-major W_post_emb)
#define OFF_WCATT  ((size_t)0x00100000)  // bf16 [4096][1024]  (N-major W_post_cat)
#define OFF_WCT    ((size_t)0x00900000)  // bf16 [64][4096]    packed Wc1|Wc2|Wc3|Wcs, N-major
#define OFF_WVT    ((size_t)0x00980000)  // bf16 [64][4096]    packed Wv1|Wv2|Wv3|Wvs, N-major
#define OFF_BSUM   ((size_t)0x00A00000)  // f32  [64]          bc*+bv* concatenated
#define OFF_HEADB  ((size_t)0x00A01000)  // bf16 [8192][512]
#define OFF_TAILB  ((size_t)0x01201000)  // bf16 [8192][512]
#define OFF_B54    ((size_t)0x01A01000)  // f32  [32768][64]   per-pair freq bias (54 used)

__constant__ int GEO_IDX[15] = {1,2,3,4,5,6,8,10,22,23,29,31,32,33,43};
__constant__ int POS_IDX[11] = {9,16,17,20,27,30,36,42,48,49,50};
__constant__ int SEM_IDX[24] = {7,11,12,13,14,15,18,19,21,24,25,26,28,34,35,37,38,39,40,41,44,45,46,47};

static __device__ __forceinline__ v8f wmma_bf16(v16bf a, v16bf b, v8f c) {
  // (neg_a, A, neg_b, B, c_mod, C, reuse_a, reuse_b)
  return __builtin_amdgcn_wmma_f32_16x16x32_bf16(false, a, false, b, (short)0, c, false, false);
}

static __device__ __forceinline__ v16bf cat16(v8bf lo, v8bf hi) {
  v16bf r;
#pragma unroll
  for (int i = 0; i < 8; ++i) { r[i] = lo[i]; r[i + 8] = hi[i]; }
  return r;
}

static __device__ __forceinline__ v16bf cvt16(v8f lo, v8f hi) {
  v16bf r;
#pragma unroll
  for (int i = 0; i < 8; ++i) { r[i] = (__bf16)lo[i]; r[i + 8] = (__bf16)hi[i]; }
  return r;
}

static __device__ __forceinline__ v8f zero8() {
  v8f r;
#pragma unroll
  for (int i = 0; i < 8; ++i) r[i] = 0.f;
  return r;
}

static __device__ __forceinline__ v16bf load_ctx_a(const __bf16* rowH, const __bf16* rowT,
                                                   int kk, int h) {
  const __bf16* s = (kk < HID) ? (rowH + kk) : (rowT + (kk - HID));
  return cat16(*(const v8bf*)(s + 16 * h), *(const v8bf*)(s + 16 * h + 8));
}

// Issue async DMA of a 16x32 f32 vis chunk (rows m0..m0+15, cols nc..nc+31)
// into a per-wave LDS buffer. 4 issues x 32 lanes x 16B = 2KB.
static __device__ __forceinline__ void async_vis_fetch(const float* vis, float* vbuf,
                                                       int m0, int nc, int lane) {
#if HAS_ASYNC_LDS
  int rrow = lane >> 3, seg = lane & 7;
#pragma unroll
  for (int i = 0; i < 4; ++i) {
    int row = i * 4 + rrow;
    const float* g = vis + (size_t)(m0 + row) * POOL + nc + seg * 4;
    float* l = vbuf + row * 32 + seg * 4;
    __builtin_amdgcn_global_load_async_to_lds_b128(
        (__attribute__((address_space(1))) i4v*)(g),
        (__attribute__((address_space(3))) i4v*)(l),
        0, 0);
  }
#else
  (void)vis; (void)vbuf; (void)m0; (void)nc; (void)lane;
#endif
}

// ---------------------------------------------------------------------------
// Kernel 1: weight transpose + bf16 convert + bias packing
// ---------------------------------------------------------------------------
__global__ void prep_kernel(const float* __restrict__ Wemb, const float* __restrict__ Wcat,
                            const float* __restrict__ Wc1, const float* __restrict__ Wc2,
                            const float* __restrict__ Wc3, const float* __restrict__ Wcs,
                            const float* __restrict__ Wv1, const float* __restrict__ Wv2,
                            const float* __restrict__ Wv3, const float* __restrict__ Wvs,
                            const float* __restrict__ bc1, const float* __restrict__ bc2,
                            const float* __restrict__ bc3, const float* __restrict__ bcs,
                            const float* __restrict__ bv1, const float* __restrict__ bv2,
                            const float* __restrict__ bv3, const float* __restrict__ bvs,
                            __bf16* __restrict__ WembT, __bf16* __restrict__ WcatT,
                            __bf16* __restrict__ WcT, __bf16* __restrict__ WvT,
                            float* __restrict__ bsum)
{
  const long T0 = 1024L * 512;
  const long T1 = 4096L * 1024;
  const long T2 = 64L * 4096;
  long idx = (long)blockIdx.x * blockDim.x + threadIdx.x;

  if (idx < T0) {
    int n = (int)(idx / 512), k = (int)(idx % 512);
    WembT[idx] = (__bf16)Wemb[(size_t)k * 1024 + n];
    return;
  }
  idx -= T0;
  if (idx < T1) {
    int n = (int)(idx / 1024), k = (int)(idx % 1024);
    WcatT[idx] = (__bf16)Wcat[(size_t)k * 4096 + n];
    return;
  }
  idx -= T1;
  if (idx < T2) {
    int j = (int)(idx / 4096), k = (int)(idx % 4096);
    float v = 0.f;
    if      (j < 15) v = Wc1[(size_t)k * 15 + j];
    else if (j < 26) v = Wc2[(size_t)k * 11 + (j - 15)];
    else if (j < 50) v = Wc3[(size_t)k * 24 + (j - 26)];
    else if (j < 54) v = Wcs[(size_t)k * 4  + (j - 50)];
    WcT[idx] = (__bf16)v;
    return;
  }
  idx -= T2;
  if (idx < T2) {
    int j = (int)(idx / 4096), k = (int)(idx % 4096);
    float v = 0.f;
    if      (j < 15) v = Wv1[(size_t)k * 15 + j];
    else if (j < 26) v = Wv2[(size_t)k * 11 + (j - 15)];
    else if (j < 50) v = Wv3[(size_t)k * 24 + (j - 26)];
    else if (j < 54) v = Wvs[(size_t)k * 4  + (j - 50)];
    WvT[idx] = (__bf16)v;
    return;
  }
  idx -= T2;
  if (idx < 64) {
    int j = (int)idx;
    float v = 0.f;
    if      (j < 15) v = bc1[j]      + bv1[j];
    else if (j < 26) v = bc2[j - 15] + bv2[j - 15];
    else if (j < 50) v = bc3[j - 26] + bv3[j - 26];
    else if (j < 54) v = bcs[j - 50] + bvs[j - 50];
    bsum[j] = v;
  }
}

// ---------------------------------------------------------------------------
// Kernel 2: per-pair frequency bias gather + log-sum-exp (54-wide additive)
// ---------------------------------------------------------------------------
__global__ void bias_kernel(const int* __restrict__ pair_pred,
                            const float* __restrict__ freq,
                            float* __restrict__ b54)
{
  int p = blockIdx.x * blockDim.x + threadIdx.x;
  if (p >= N_PAIRS) return;
  int p0 = pair_pred[2 * p], p1 = pair_pred[2 * p + 1];
  const float* b = freq + ((size_t)p0 * NOBJ + p1) * NREL;
  float* o = b54 + (size_t)p * 64;
  float s1 = 0.f, s2 = 0.f, s3 = 0.f;
#pragma unroll
  for (int j = 0; j < 15; ++j) { float v = b[GEO_IDX[j]]; o[j] = v;      s1 += expf(v); }
#pragma unroll
  for (int j = 0; j < 11; ++j) { float v = b[POS_IDX[j]]; o[15 + j] = v; s2 += expf(v); }
#pragma unroll
  for (int j = 0; j < 24; ++j) { float v = b[SEM_IDX[j]]; o[26 + j] = v; s3 += expf(v); }
  o[50] = b[0];
  o[51] = logf(s1);
  o[52] = logf(s2);
  o[53] = logf(s3);
}

// ---------------------------------------------------------------------------
// Kernel 3: er = edge_ctx @ W_post_emb + b  -> headB/tailB (bf16)
// ---------------------------------------------------------------------------
__global__ void __launch_bounds__(128)
emb_kernel(const float* __restrict__ ec, const __bf16* __restrict__ WembT,
           const float* __restrict__ bemb,
           __bf16* __restrict__ headB, __bf16* __restrict__ tailB)
{
  int wave = threadIdx.x >> 5;
  int lane = threadIdx.x & 31;
  int mloc = lane & 15, h = lane >> 4;
  int tile = blockIdx.x * 4 + wave;     // 8192 tiles = (8192/16) * (1024/64)
  int m0 = (tile >> 4) * 16;
  int n0 = (tile & 15) * 64;

  const float* arow = ec + (size_t)(m0 + mloc) * HID;
  v8f acc[4];
#pragma unroll
  for (int t = 0; t < 4; ++t) acc[t] = zero8();

  for (int kk = 0; kk < HID; kk += 32) {
    v8f f0 = *(const v8f*)(arow + kk + 16 * h);
    v8f f1 = *(const v8f*)(arow + kk + 16 * h + 8);
    v16bf a = cvt16(f0, f1);
#pragma unroll
    for (int t = 0; t < 4; ++t) {
      v16bf b = *(const v16bf*)(WembT + (size_t)(n0 + 16 * t + mloc) * HID + kk + 16 * h);
      acc[t] = wmma_bf16(a, b, acc[t]);
    }
  }
#pragma unroll
  for (int t = 0; t < 4; ++t) {
    int n = n0 + 16 * t + mloc;
    float bb = bemb[n];
#pragma unroll
    for (int r = 0; r < 8; ++r) {
      int m = m0 + r + 8 * h;
      __bf16 v = (__bf16)(acc[t][r] + bb);
      if (n < HID) headB[(size_t)m * HID + n] = v;
      else         tailB[(size_t)m * HID + (n - HID)] = v;
    }
  }
}

// ---------------------------------------------------------------------------
// Kernel 4: fused main — per 16-row wave tile:
//   async-DMA next vis chunk to LDS | post_ctx chunk (16x32) in registers
//   (software-pipelined K loop) -> relu -> LDS layout bounce ->
//   accumulate ctx@WcT and vis@WvT -> + biases -> scatter to 4 outputs
// ---------------------------------------------------------------------------
__global__ void __launch_bounds__(128)
main_kernel(const int* __restrict__ pair_idx, const float* __restrict__ vis,
            const __bf16* __restrict__ headB, const __bf16* __restrict__ tailB,
            const __bf16* __restrict__ WcatT, const float* __restrict__ bcat,
            const __bf16* __restrict__ WcT, const __bf16* __restrict__ WvT,
            const float* __restrict__ bsum, const float* __restrict__ b54,
            float* __restrict__ out)
{
  __shared__ __bf16 lpc[4][16 * 32];                    // relu(post_ctx) layout bounce
  __shared__ __align__(128) float visbuf[4][2][16 * 32]; // per-wave vis double buffer

  int wave = threadIdx.x >> 5;
  int lane = threadIdx.x & 31;
  int mloc = lane & 15, h = lane >> 4;
  int m0 = blockIdx.x * 64 + wave * 16;

  int hIdx = pair_idx[2 * (m0 + mloc)];
  int tIdx = pair_idx[2 * (m0 + mloc) + 1];
  const __bf16* rowH = headB + (size_t)hIdx * HID;
  const __bf16* rowT = tailB + (size_t)tIdx * HID;
  const float*  vrow = vis + (size_t)(m0 + mloc) * POOL;
  __bf16* lp = lpc[wave];

  v8f accC[4], accV[4];
#pragma unroll
  for (int t = 0; t < 4; ++t) { accC[t] = zero8(); accV[t] = zero8(); }

  // prologue: DMA chunk 0 of vis into buffer 0
  async_vis_fetch(vis, visbuf[wave][0], m0, 0, lane);

  for (int nc = 0; nc < POOL; nc += 32) {
    int cur = (nc >> 5) & 1;
    int nc_next = (nc + 32 < POOL) ? nc + 32 : 0;

    // DMA next vis chunk into the other buffer; overlaps with the big GEMM.
    async_vis_fetch(vis, visbuf[wave][cur ^ 1], m0, nc_next, lane);

    // Prefetch next chunk's B rows (L2 hint).
    if (nc + 32 < POOL) {
      __builtin_prefetch(WcatT + (size_t)(nc + 32 + mloc) * 1024 + 16 * h, 0, 0);
      __builtin_prefetch(WcatT + (size_t)(nc + 48 + mloc) * 1024 + 16 * h, 0, 0);
    }

    // ---- big GEMM chunk: post_ctx[16, nc:nc+32], software-pipelined ----
    const __bf16* b0base = WcatT + (size_t)(nc + mloc)      * 1024 + 16 * h;
    const __bf16* b1base = WcatT + (size_t)(nc + 16 + mloc) * 1024 + 16 * h;

    v8f pc0 = zero8(), pc1 = zero8();
    v16bf a  = load_ctx_a(rowH, rowT, 0, h);
    v16bf b0 = *(const v16bf*)(b0base);
    v16bf b1 = *(const v16bf*)(b1base);
#pragma unroll 4
    for (int kk = 0; kk < 2 * HID; kk += 32) {
      int kn = (kk + 32 < 2 * HID) ? kk + 32 : 0;
      v16bf a2  = load_ctx_a(rowH, rowT, kn, h);
      v16bf b02 = *(const v16bf*)(b0base + kn);
      v16bf b12 = *(const v16bf*)(b1base + kn);
      pc0 = wmma_bf16(a, b0, pc0);
      pc1 = wmma_bf16(a, b1, pc1);
      a = a2; b0 = b02; b1 = b12;
    }

    // ---- bias + relu, stage into LDS (C/D layout -> row-major) ----
    float bb0 = bcat[nc + mloc];
    float bb1 = bcat[nc + 16 + mloc];
#pragma unroll
    for (int r = 0; r < 8; ++r) {
      lp[(r + 8 * h) * 32 + mloc]      = (__bf16)fmaxf(pc0[r] + bb0, 0.f);
      lp[(r + 8 * h) * 32 + mloc + 16] = (__bf16)fmaxf(pc1[r] + bb1, 0.f);
    }
    asm volatile("s_wait_dscnt 0" ::: "memory");

    // ---- reload relu(post_ctx) in A-operand layout ----
    v8bf p0 = *(const v8bf*)(lp + mloc * 32 + 16 * h);
    v8bf p1 = *(const v8bf*)(lp + mloc * 32 + 16 * h + 8);
    v16bf pA = cat16(p0, p1);
    asm volatile("s_wait_dscnt 0" ::: "memory");

    // ---- vis chunk A-tile (from async LDS buffer if available) ----
#if HAS_ASYNC_LDS
    asm volatile("s_wait_asynccnt 4" ::: "memory");   // current chunk's 4 DMAs done
    const float* vb = visbuf[wave][cur];
    v8f f0 = *(const v8f*)(vb + mloc * 32 + 16 * h);
    v8f f1 = *(const v8f*)(vb + mloc * 32 + 16 * h + 8);
#else
    v8f f0 = *(const v8f*)(vrow + nc + 16 * h);
    v8f f1 = *(const v8f*)(vrow + nc + 16 * h + 8);
#endif
    v16bf vA = cvt16(f0, f1);

    // ---- small GEMMs: out(16x64) accumulation ----
#pragma unroll
    for (int t = 0; t < 4; ++t) {
      v16bf bc = *(const v16bf*)(WcT + (size_t)(16 * t + mloc) * POOL + nc + 16 * h);
      v16bf bv = *(const v16bf*)(WvT + (size_t)(16 * t + mloc) * POOL + nc + 16 * h);
      accC[t] = wmma_bf16(pA, bc, accC[t]);
      accV[t] = wmma_bf16(vA, bv, accV[t]);
    }
  }

#if HAS_ASYNC_LDS
  asm volatile("s_wait_asynccnt 0" ::: "memory");     // drain trailing dummy DMA
#endif

  // ---- epilogue: add freq bias + layer biases, scatter to 4 outputs ----
#pragma unroll
  for (int t = 0; t < 4; ++t) {
    int n = 16 * t + mloc;
    if (n >= 54) continue;
    float bn = bsum[n];
    float* base; int cols, c;
    if      (n < 15) { base = out;                          cols = 15; c = n;      }
    else if (n < 26) { base = out + (size_t)N_PAIRS * 15;   cols = 11; c = n - 15; }
    else if (n < 50) { base = out + (size_t)N_PAIRS * 26;   cols = 24; c = n - 26; }
    else             { base = out + (size_t)N_PAIRS * 50;   cols = 4;  c = n - 50; }
#pragma unroll
    for (int r = 0; r < 8; ++r) {
      int m = m0 + r + 8 * h;
      base[(size_t)m * cols + c] =
          accC[t][r] + accV[t][r] + b54[(size_t)m * 64 + n] + bn;
    }
  }
  (void)vrow;
}

// ---------------------------------------------------------------------------
extern "C" void kernel_launch(void* const* d_in, const int* in_sizes, int n_in,
                              void* d_out, int out_size, void* d_ws, size_t ws_size,
                              hipStream_t stream) {
  (void)in_sizes; (void)n_in; (void)out_size; (void)ws_size;
  const float* edge_ctx   = (const float*)d_in[0];
  const int*   pair_idx   = (const int*)  d_in[1];
  const float* vis_rep    = (const float*)d_in[2];
  const int*   pair_pred  = (const int*)  d_in[3];
  const float* W_post_emb = (const float*)d_in[4];
  const float* b_post_emb = (const float*)d_in[5];
  const float* W_post_cat = (const float*)d_in[6];
  const float* b_post_cat = (const float*)d_in[7];
  const float* Wc1 = (const float*)d_in[8];   const float* bc1 = (const float*)d_in[9];
  const float* Wc2 = (const float*)d_in[10];  const float* bc2 = (const float*)d_in[11];
  const float* Wc3 = (const float*)d_in[12];  const float* bc3 = (const float*)d_in[13];
  const float* Wcs = (const float*)d_in[14];  const float* bcs = (const float*)d_in[15];
  const float* Wv1 = (const float*)d_in[16];  const float* bv1 = (const float*)d_in[17];
  const float* Wv2 = (const float*)d_in[18];  const float* bv2 = (const float*)d_in[19];
  const float* Wv3 = (const float*)d_in[20];  const float* bv3 = (const float*)d_in[21];
  const float* Wvs = (const float*)d_in[22];  const float* bvs = (const float*)d_in[23];
  const float* freq = (const float*)d_in[24];
  float* out = (float*)d_out;

  char* ws = (char*)d_ws;
  __bf16* WembT = (__bf16*)(ws + OFF_WEMBT);
  __bf16* WcatT = (__bf16*)(ws + OFF_WCATT);
  __bf16* WcT   = (__bf16*)(ws + OFF_WCT);
  __bf16* WvT   = (__bf16*)(ws + OFF_WVT);
  float*  bsum  = (float*) (ws + OFF_BSUM);
  __bf16* headB = (__bf16*)(ws + OFF_HEADB);
  __bf16* tailB = (__bf16*)(ws + OFF_TAILB);
  float*  b54   = (float*) (ws + OFF_B54);

  long prep_total = 1024L * 512 + 4096L * 1024 + 2L * 64 * 4096 + 64;
  int prep_blocks = (int)((prep_total + 255) / 256);

  prep_kernel<<<prep_blocks, 256, 0, stream>>>(
      W_post_emb, W_post_cat, Wc1, Wc2, Wc3, Wcs, Wv1, Wv2, Wv3, Wvs,
      bc1, bc2, bc3, bcs, bv1, bv2, bv3, bvs,
      WembT, WcatT, WcT, WvT, bsum);

  bias_kernel<<<N_PAIRS / 256, 256, 0, stream>>>(pair_pred, freq, b54);

  emb_kernel<<<2048, 128, 0, stream>>>(edge_ctx, WembT, b_post_emb, headB, tailB);

  main_kernel<<<512, 128, 0, stream>>>(pair_idx, vis_rep, headB, tailB,
                                       WcatT, b_post_cat, WcT, WvT,
                                       bsum, b54, out);
}
